// FullDPM_83219286327659
// MI455X (gfx1250) — compile-verified, matrix operand order checked by reading, not run on hardware
//
#include <hip/hip_runtime.h>
#include <hip/hip_bf16.h>

// ---------------- problem constants ----------------
#define NB   4
#define LL   512
#define DD   128
#define PDIM 64
#define NLAY 6
#define HH   12
#define DKH  32
#define PP   8
#define NF4  256
#define TOK  (NB*LL)          // 2048
#define PPAD 32               // padded point dim per (tok,h): 24 -> 32
#define INFP 416              // padded head input dim: 387 -> 416

typedef __attribute__((ext_vector_type(16))) __bf16 v16bf;
typedef __attribute__((ext_vector_type(8)))  float  v8f;

struct GemmP {
  const float* A; const float* B; const float* bias;
  const unsigned char* rmask;
  float* C;
  int M, Ncols, K;
  long lda, ldb, ldc;
  long sAn, sAh, sBn, sBh, sCn, sCh;
  int Hdiv, relu, transC;
};

static __device__ inline v16bf pack16(float4 a0, float4 a1, float4 a2, float4 a3) {
  v16bf v;
  v[0]=(__bf16)a0.x; v[1]=(__bf16)a0.y; v[2]=(__bf16)a0.z; v[3]=(__bf16)a0.w;
  v[4]=(__bf16)a1.x; v[5]=(__bf16)a1.y; v[6]=(__bf16)a1.z; v[7]=(__bf16)a1.w;
  v[8]=(__bf16)a2.x; v[9]=(__bf16)a2.y; v[10]=(__bf16)a2.z; v[11]=(__bf16)a2.w;
  v[12]=(__bf16)a3.x; v[13]=(__bf16)a3.y; v[14]=(__bf16)a3.z; v[15]=(__bf16)a3.w;
  return v;
}

// ---------- fast GEMM: M%16==0, (N/16)%NACC==0, K%32==0, 16B-aligned A (and B if TRANSB) ----
// wave computes a 16x(16*NACC) C tile; block = 4 waves. Fully static inner loop.
template<int NACC, int TRANSB>
__global__ __launch_bounds__(128) void k_gemm_fast(GemmP g) {
  int wave = threadIdx.x >> 5;
  int lane = threadIdx.x & 31;
  int half = lane >> 4;
  int mr   = lane & 15;
  int tile_m = blockIdx.x * 64 + wave * 16;
  if (tile_m >= g.M) return;                 // wave-uniform
  int tile_n = blockIdx.y * (16 * NACC);
  int z = blockIdx.z;
  int zn = z / g.Hdiv, zh = z - zn * g.Hdiv;
  const float* __restrict__ A = g.A + zn*g.sAn + zh*g.sAh;
  const float* __restrict__ B = g.B + zn*g.sBn + zh*g.sBh;
  float*       __restrict__ C = g.C + zn*g.sCn + zh*g.sCh;

  int rowA = tile_m + mr;
  int kb   = half * 8;                       // ISA 16-bit operand K split per lane half

  v8f acc[NACC] = {};
  const float* arow = A + (long)rowA * g.lda + kb;

  for (int k0 = 0; k0 < g.K; k0 += 32) {
    float4 a0 = *(const float4*)(arow + k0);
    float4 a1 = *(const float4*)(arow + k0 + 4);
    float4 a2 = *(const float4*)(arow + k0 + 16);
    float4 a3 = *(const float4*)(arow + k0 + 20);
    v16bf av = pack16(a0, a1, a2, a3);

    v16bf bfr[NACC];
    if (TRANSB) {
#pragma unroll
      for (int c = 0; c < NACC; ++c) {
        const float* bcol = B + (long)(tile_n + c*16 + mr) * g.ldb + kb + k0;
        bfr[c] = pack16(*(const float4*)(bcol),
                        *(const float4*)(bcol + 4),
                        *(const float4*)(bcol + 16),
                        *(const float4*)(bcol + 20));
      }
    } else {
#pragma unroll
      for (int c = 0; c < NACC; ++c) {
        const float* bp = B + (long)k0 * g.ldb + tile_n + c*16 + mr;
        float tmp[16];
#pragma unroll
        for (int e = 0; e < 8; ++e) tmp[e]   = bp[(long)(kb+e)    * g.ldb];
#pragma unroll
        for (int e = 0; e < 8; ++e) tmp[8+e] = bp[(long)(16+kb+e) * g.ldb];
        v16bf bb;
#pragma unroll
        for (int e = 0; e < 16; ++e) bb[e] = (__bf16)tmp[e];
        bfr[c] = bb;
      }
    }
#pragma unroll
    for (int c = 0; c < NACC; ++c)
      acc[c] = __builtin_amdgcn_wmma_f32_16x16x32_bf16(false, av, false, bfr[c],
                                                       (short)0, acc[c], false, false);
  }

#pragma unroll
  for (int c = 0; c < NACC; ++c) {
    int col = tile_n + c*16 + mr;
    float bb = g.bias ? g.bias[col] : 0.f;
#pragma unroll
    for (int r = 0; r < 8; ++r) {
      int row = tile_m + r + half*8;
      float vv = acc[c][r] + bb;
      if (g.relu) vv = fmaxf(vv, 0.f);
      if (g.rmask && !g.rmask[row]) vv = 0.f;
      if (g.transC) C[(long)col * g.ldc + row] = vv;
      else          C[(long)row * g.ldc + col] = vv;
    }
  }
}

// ---------------- transpose + zero-pad weights: out(Npad,Kpad) from in(K,N) ----------------
__global__ void k_tr(const float* in, float* out, int K, int N, int Kpad, int Npad) {
  int t = blockIdx.x * blockDim.x + threadIdx.x;
  if (t >= Npad * Kpad) return;
  int n = t / Kpad, k = t % Kpad;
  out[t] = (k < K && n < N) ? in[(long)k * N + n] : 0.f;
}

// ---------------- elementwise kernels ----------------
__global__ void k_prep(const float* v_t, const float* res, const int* s_t,
                       const float* embed, float* Rb, float* hb) {
  int t = blockIdx.x * blockDim.x + threadIdx.x;
  if (t >= TOK) return;
  float vx = v_t[t*3], vy = v_t[t*3+1], vz = v_t[t*3+2];
  float th = sqrtf(vx*vx + vy*vy + vz*vz + 1e-12f);
  float ux = vx/th, uy = vy/th, uz = vz/th;
  float s = sinf(th), c = cosf(th), oc = 1.f - c;
  float R[9];
  R[0]=c+oc*ux*ux;     R[1]=oc*ux*uy-s*uz; R[2]=oc*ux*uz+s*uy;
  R[3]=oc*ux*uy+s*uz;  R[4]=c+oc*uy*uy;    R[5]=oc*uy*uz-s*ux;
  R[6]=oc*ux*uz-s*uy;  R[7]=oc*uy*uz+s*ux; R[8]=c+oc*uz*uz;
  for (int k = 0; k < 9; ++k) Rb[t*9+k] = R[k];
  int si = s_t[t];
  for (int k = 0; k < DD; ++k) {
    hb[(long)t*2*DD + k]      = res[(long)t*DD + k];
    hb[(long)t*2*DD + DD + k] = embed[(long)si*DD + k];
  }
}

// rotate+translate 8 points of one (tok,h); row layout (tok,h,PPAD) or transposed
// layout [(h*32+idx)*TOK + tok]; zero-pads idx 24..31; optional sq-norm output.
__global__ void k_rotsq(const float* Rb, const float* praw, const float* p_t,
                        float* pg, float* sq, int transposed) {
  int t = blockIdx.x * blockDim.x + threadIdx.x;   // over TOK*H
  if (t >= TOK*HH) return;
  int h = t % HH, tok = t / HH;
  const float* R = Rb + tok*9;
  float px = p_t[tok*3], py = p_t[tok*3+1], pz = p_t[tok*3+2];
  const float* in = praw + (long)tok*(HH*PP*3) + h*24;
  float s = 0.f;
  for (int p = 0; p < PP; ++p) {
    float x = in[p*3], y = in[p*3+1], z = in[p*3+2];
    float gx = R[0]*x + R[1]*y + R[2]*z + px;
    float gy = R[3]*x + R[4]*y + R[5]*z + py;
    float gz = R[6]*x + R[7]*y + R[8]*z + pz;
    if (transposed) {
      pg[(long)(h*PPAD + p*3    )*TOK + tok] = gx;
      pg[(long)(h*PPAD + p*3 + 1)*TOK + tok] = gy;
      pg[(long)(h*PPAD + p*3 + 2)*TOK + tok] = gz;
    } else {
      float* o = pg + (long)tok*(HH*PPAD) + h*PPAD;
      o[p*3] = gx; o[p*3+1] = gy; o[p*3+2] = gz;
    }
    s += gx*gx + gy*gy + gz*gz;
  }
  if (transposed) {
    for (int k = 24; k < PPAD; ++k) pg[(long)(h*PPAD + k)*TOK + tok] = 0.f;
  } else {
    float* o = pg + (long)tok*(HH*PPAD) + h*PPAD;
    for (int k = 24; k < PPAD; ++k) o[k] = 0.f;
  }
  if (sq) sq[t] = s;
}

__global__ __launch_bounds__(256) void k_softmax(float* S, const float* Cr,
    const float* biasb, const float* sqq, const float* sqk,
    const float* gamma, const unsigned char* mres, int layer) {
  __shared__ float red[256];
  int b = blockIdx.x;                       // n*H*L + h*L + i
  int i = b % LL; int nh = b / LL; int h = nh % HH; int n = nh / HH;
  int t = threadIdx.x;
  float g  = gamma[layer*HH + h];
  float gs = logf(1.f + expf(g));           // softplus
  const float inv_sdk = 0.17677669529663687f;  // 1/sqrt(32)
  const float inv_s3  = 0.5773502691896258f;   // 1/sqrt(3)
  const float wc = 1.f/6.f;                    // sqrt(2/(9*8))
  long rowoff = (((long)(n*HH + h) * LL) + i) * LL;
  float sqi = sqq[(n*LL + i)*HH + h];
  bool  mi  = mres[n*LL + i] != 0;
  float vals[2];
#pragma unroll
  for (int u = 0; u < 2; ++u) {
    int j = t + u*256;
    float qk = S[rowoff + j];
    float cr = Cr[rowoff + j];
    float bi = biasb[((long)(n*LL + i)*LL + j)*16 + h];
    float d2 = sqi + sqk[(n*LL + j)*HH + h] - 2.f*cr;
    float lg = (qk*inv_sdk + bi - 0.5f*wc*gs*d2) * inv_s3;
    bool mj = mres[n*LL + j] != 0;
    if (!(mi && mj)) lg = -1e9f;
    vals[u] = lg;
  }
  float m = fmaxf(vals[0], vals[1]);
  red[t] = m; __syncthreads();
  for (int o = 128; o > 0; o >>= 1) { if (t < o) red[t] = fmaxf(red[t], red[t+o]); __syncthreads(); }
  m = red[0]; __syncthreads();
  float e0 = expf(vals[0]-m), e1 = expf(vals[1]-m);
  red[t] = e0 + e1; __syncthreads();
  for (int o = 128; o > 0; o >>= 1) { if (t < o) red[t] += red[t+o]; __syncthreads(); }
  float inv = 1.f / red[0];
  S[rowoff + t]       = e0 * inv;
  S[rowoff + t + 256] = e1 * inv;
}

__global__ void k_feat(const float* o_s, const float* o_p, const float* o_ptg,
                       const float* Rb, const float* p_t, float* feat) {
  int t = blockIdx.x * blockDim.x + threadIdx.x;   // over TOK*H
  if (t >= TOK*HH) return;
  int h = t % HH, tok = t / HH;
  float* f = feat + (long)tok*1536;
  const float* os = o_s + (long)tok*(HH*DKH) + h*DKH;
  for (int d = 0; d < DKH; ++d) f[h*DKH + d] = os[d];
  const float* op = o_p + (long)tok*(16*PDIM) + h*PDIM;   // padded M=16 layout
  for (int d = 0; d < PDIM; ++d) f[384 + h*PDIM + d] = op[d];
  const float* R = Rb + tok*9;
  float px = p_t[tok*3], py = p_t[tok*3+1], pz = p_t[tok*3+2];
  const float* og = o_ptg + (long)tok*(HH*PPAD) + h*PPAD;
  for (int p = 0; p < PP; ++p) {
    float dx = og[p*3]-px, dy = og[p*3+1]-py, dz = og[p*3+2]-pz;
    float ox = R[0]*dx + R[3]*dy + R[6]*dz;   // R^T * d
    float oy = R[1]*dx + R[4]*dy + R[7]*dz;
    float oz = R[2]*dx + R[5]*dy + R[8]*dz;
    f[1152 + h*24 + p*3 + 0] = ox;
    f[1152 + h*24 + p*3 + 1] = oy;
    f[1152 + h*24 + p*3 + 2] = oz;
    f[1440 + h*PP + p] = sqrtf(ox*ox + oy*oy + oz*oz + 1e-8f);
  }
}

__global__ __launch_bounds__(128) void k_addln(float* x, const float* d,
                                               const float* gw, const float* bw) {
  __shared__ float s[128];
  int tok = blockIdx.x; int i = threadIdx.x;
  float v = x[(long)tok*DD + i] + d[(long)tok*DD + i];
  s[i] = v; __syncthreads();
  for (int o = 64; o > 0; o >>= 1) { if (i < o) s[i] += s[i+o]; __syncthreads(); }
  float m = s[0] / 128.f; __syncthreads();
  float c = v - m; s[i] = c*c; __syncthreads();
  for (int o = 64; o > 0; o >>= 1) { if (i < o) s[i] += s[i+o]; __syncthreads(); }
  float var = s[0] / 128.f;
  x[(long)tok*DD + i] = c * rsqrtf(var + 1e-5f) * gw[i] + bw[i];
}

__global__ void k_infeat(const float* tree, const float* x, const float* beta,
                         float* inf) {
  int t = blockIdx.x * blockDim.x + threadIdx.x;
  if (t >= TOK) return;
  float* o = inf + (long)t*INFP;
  for (int k = 0; k < NF4; ++k) o[k] = tree[(long)t*NF4 + k];
  for (int k = 0; k < DD;  ++k) o[NF4 + k] = x[(long)t*DD + k];
  float b = beta[t / LL];
  o[384] = b; o[385] = sinf(b); o[386] = cosf(b);
  for (int k = 387; k < INFP; ++k) o[k] = 0.f;
}

__global__ void k_final(const float* Rb, const float* v_t,
                        const float* o_crd, const float* o_rot, const float* o_seq,
                        const float* b_crd, const float* b_rot, const float* b_seq,
                        const unsigned char* mgen, float* out) {
  int t = blockIdx.x * blockDim.x + threadIdx.x;
  if (t >= TOK) return;
  float qx = o_rot[t*32+0] + b_rot[0];
  float qy = o_rot[t*32+1] + b_rot[1];
  float qz = o_rot[t*32+2] + b_rot[2];
  float qn = rsqrtf(1.f + qx*qx + qy*qy + qz*qz);
  float w = qn, x = qx*qn, y = qy*qn, z = qz*qn;
  float U[9] = {
    1-2*(y*y+z*z), 2*(x*y-w*z), 2*(x*z+w*y),
    2*(x*y+w*z), 1-2*(x*x+z*z), 2*(y*z-w*x),
    2*(x*z-w*y), 2*(y*z+w*x), 1-2*(x*x+y*y)};
  const float* R = Rb + t*9;
  float Rn[9];
  for (int i = 0; i < 3; ++i)
    for (int j = 0; j < 3; ++j) {
      float s = 0.f;
      for (int k = 0; k < 3; ++k) s += R[i*3+k] * U[k*3+j];
      Rn[i*3+j] = s;
    }
  for (int k = 0; k < 9; ++k) out[TOK*3 + t*9 + k] = Rn[k];   // R_next
  float tr = Rn[0] + Rn[4] + Rn[8];
  float cs = fminf(fmaxf((tr-1.f)*0.5f, -1.f+1e-6f), 1.f-1e-6f);
  float th = acosf(cs);
  float s2 = 2.f * sinf(th);
  float w0 = Rn[7]-Rn[5], w1 = Rn[2]-Rn[6], w2 = Rn[3]-Rn[1];
  bool m = mgen[t] != 0;
  out[t*3+0] = m ? w0/s2*th : v_t[t*3+0];
  out[t*3+1] = m ? w1/s2*th : v_t[t*3+1];
  out[t*3+2] = m ? w2/s2*th : v_t[t*3+2];
  long epo = (long)TOK*3 + (long)TOK*9;
  for (int c = 0; c < 3; ++c)
    out[epo + t*3 + c] = m ? (o_crd[t*32+c] + b_crd[c]) : 0.f;
  long sqo = epo + (long)TOK*3;
  float mx = -1e30f;
  float lv[20];
  for (int k = 0; k < 20; ++k) { lv[k] = o_seq[t*32+k] + b_seq[k]; mx = fmaxf(mx, lv[k]); }
  float sm = 0.f;
  for (int k = 0; k < 20; ++k) sm += expf(lv[k] - mx);
  float inv = 1.f / sm;
  for (int k = 0; k < 20; ++k) out[sqo + t*20 + k] = expf(lv[k] - mx) * inv;
}

// ---------------- host side ----------------
static void gemmF(hipStream_t s, const float* A, const float* B, const float* bias,
                  const unsigned char* rmask, float* C, int M, int N, int K,
                  long lda, long ldb, long ldc,
                  long sAn, long sAh, long sBn, long sBh, long sCn, long sCh,
                  int nz, int Hdiv, int transB, int relu, int transC = 0) {
  GemmP g{A,B,bias,rmask,C,M,N,K,lda,ldb,ldc,sAn,sAh,sBn,sBh,sCn,sCh,Hdiv,relu,transC};
  int tiles = N / 16;
  int nacc = (tiles % 4 == 0) ? 4 : (tiles % 2 == 0) ? 2 : 1;
  dim3 grid((M + 63) / 64, tiles / nacc, nz);
  if (transB) {
    if (nacc == 4)      k_gemm_fast<4,1><<<grid, 128, 0, s>>>(g);
    else if (nacc == 2) k_gemm_fast<2,1><<<grid, 128, 0, s>>>(g);
    else                k_gemm_fast<1,1><<<grid, 128, 0, s>>>(g);
  } else {
    k_gemm_fast<4,0><<<grid, 128, 0, s>>>(g);   // only used with N=64
  }
}

static void trW(hipStream_t s, const float* in, float* out, int K, int N,
                int Kpad, int Npad) {
  int tot = Npad * Kpad;
  k_tr<<<(tot + 255) / 256, 256, 0, s>>>(in, out, K, N, Kpad, Npad);
}

extern "C" void kernel_launch(void* const* d_in, const int* in_sizes, int n_in,
                              void* d_out, int out_size, void* d_ws, size_t ws_size,
                              hipStream_t stream) {
  const float* v_t  = (const float*)d_in[0];
  const float* p_t  = (const float*)d_in[1];
  const int*   s_t  = (const int*)d_in[2];
  const float* res  = (const float*)d_in[3];
  const float* pair = (const float*)d_in[4];
  const float* beta = (const float*)d_in[5];
  const unsigned char* mgen = (const unsigned char*)d_in[6];
  const unsigned char* mres = (const unsigned char*)d_in[7];
  const float* tree = (const float*)d_in[8];
  const float* embed  = (const float*)d_in[9];
  const float* mix_w1 = (const float*)d_in[10];
  const float* mix_b1 = (const float*)d_in[11];
  const float* mix_w2 = (const float*)d_in[12];
  const float* mix_b2 = (const float*)d_in[13];
  const float* Wq  = (const float*)d_in[14];
  const float* Wk  = (const float*)d_in[15];
  const float* Wv  = (const float*)d_in[16];
  const float* Wb  = (const float*)d_in[17];
  const float* Wpq = (const float*)d_in[18];
  const float* Wpk = (const float*)d_in[19];
  const float* Wpv = (const float*)d_in[20];
  const float* gamma = (const float*)d_in[21];
  const float* Wo  = (const float*)d_in[22];
  const float* bo  = (const float*)d_in[23];
  const float* ln1g = (const float*)d_in[24];
  const float* ln1b = (const float*)d_in[25];
  const float* Wt1 = (const float*)d_in[26];
  const float* bt1 = (const float*)d_in[27];
  const float* Wt2 = (const float*)d_in[28];
  const float* bt2 = (const float*)d_in[29];
  const float* ln2g = (const float*)d_in[30];
  const float* ln2b = (const float*)d_in[31];
  const float* head_w[3][3] = {
    {(const float*)d_in[32], (const float*)d_in[34], (const float*)d_in[36]},
    {(const float*)d_in[38], (const float*)d_in[40], (const float*)d_in[42]},
    {(const float*)d_in[44], (const float*)d_in[46], (const float*)d_in[48]}};
  const float* head_b[3][3] = {
    {(const float*)d_in[33], (const float*)d_in[35], (const float*)d_in[37]},
    {(const float*)d_in[39], (const float*)d_in[41], (const float*)d_in[43]},
    {(const float*)d_in[45], (const float*)d_in[47], (const float*)d_in[49]}};

  float* W = (float*)d_ws;
  auto alloc = [&](size_t n) { float* p = W; W += n; return p; };
  float* Rb   = alloc((size_t)TOK*9);
  float* hb   = alloc((size_t)TOK*2*DD);
  float* mixh = alloc((size_t)TOK*DD);
  float* xb   = alloc((size_t)TOK*DD);
  float* qb   = alloc((size_t)TOK*HH*DKH);
  float* kbuf = alloc((size_t)TOK*HH*DKH);
  float* vT   = alloc((size_t)HH*DKH*TOK);      // transposed V: [(h*32+d)*TOK + tok]
  float* praw = alloc((size_t)TOK*HH*PP*3);
  float* qpg  = alloc((size_t)TOK*HH*PPAD);
  float* kpg  = alloc((size_t)TOK*HH*PPAD);
  float* vpgT = alloc((size_t)HH*PPAD*TOK);     // transposed value points
  float* sqq  = alloc((size_t)TOK*HH);
  float* sqk  = alloc((size_t)TOK*HH);
  float* biasb= alloc((size_t)NB*LL*LL*16);
  float* Sb   = alloc((size_t)NB*HH*LL*LL);
  float* Crb  = alloc((size_t)NB*HH*LL*LL);
  float* osb  = alloc((size_t)TOK*HH*DKH);
  float* opb  = alloc((size_t)TOK*16*PDIM);     // padded M=16
  float* optg = alloc((size_t)TOK*HH*PPAD);
  float* featb= alloc((size_t)TOK*1536);
  float* f1   = alloc((size_t)TOK*DD);
  float* f2   = alloc((size_t)TOK*DD);
  float* infb = alloc((size_t)TOK*INFP);
  float* h1   = alloc((size_t)TOK*256);
  float* h2   = alloc((size_t)TOK*128);
  float* ocrd = alloc((size_t)TOK*32);
  float* orot = alloc((size_t)TOK*32);
  float* oseq = alloc((size_t)TOK*32);
  float* wtA  = alloc((size_t)1536*128);        // scratch for transposed weights (max Wo)
  float* WbT  = alloc((size_t)16*64);

  k_prep<<<(TOK+63)/64, 64, 0, stream>>>(v_t, res, s_t, embed, Rb, hb);
  trW(stream, mix_w1, wtA, 2*DD, DD, 2*DD, DD);
  gemmF(stream, hb, wtA, mix_b1, nullptr, mixh, TOK, DD, 2*DD, 2*DD, 2*DD, DD,
        0,0,0,0,0,0, 1,1, 1,1);
  trW(stream, mix_w2, wtA, DD, DD, DD, DD);
  gemmF(stream, mixh, wtA, mix_b2, nullptr, xb, TOK, DD, DD, DD, DD, DD,
        0,0,0,0,0,0, 1,1, 1,0);

  const long sQn = (long)LL*HH*DKH;        // per-n stride for q/k (row layout)
  const long sPn = (long)LL*HH*PPAD;       // per-n stride for padded points (row layout)
  const long sSn = (long)HH*LL*LL;         // per-n stride for logits

  for (int l = 0; l < NLAY; ++l) {
    const float* Wq_l  = Wq  + (size_t)l*DD*HH*DKH;
    const float* Wk_l  = Wk  + (size_t)l*DD*HH*DKH;
    const float* Wv_l  = Wv  + (size_t)l*DD*HH*DKH;
    const float* Wb_l  = Wb  + (size_t)l*PDIM*HH;
    const float* Wpq_l = Wpq + (size_t)l*DD*HH*PP*3;
    const float* Wpk_l = Wpk + (size_t)l*DD*HH*PP*3;
    const float* Wpv_l = Wpv + (size_t)l*DD*HH*PP*3;
    const float* Wo_l  = Wo  + (size_t)l*1536*DD;

    trW(stream, Wq_l, wtA, DD, HH*DKH, DD, HH*DKH);
    gemmF(stream, xb, wtA, nullptr, nullptr, qb, TOK, HH*DKH, DD, DD, DD, HH*DKH,
          0,0,0,0,0,0, 1,1, 1,0);
    trW(stream, Wk_l, wtA, DD, HH*DKH, DD, HH*DKH);
    gemmF(stream, xb, wtA, nullptr, nullptr, kbuf, TOK, HH*DKH, DD, DD, DD, HH*DKH,
          0,0,0,0,0,0, 1,1, 1,0);
    trW(stream, Wv_l, wtA, DD, HH*DKH, DD, HH*DKH);
    gemmF(stream, xb, wtA, nullptr, nullptr, vT, TOK, HH*DKH, DD, DD, DD, TOK,
          0,0,0,0,0,0, 1,1, 1,0, /*transC*/1);

    trW(stream, Wpq_l, wtA, DD, HH*PP*3, DD, HH*PP*3);
    gemmF(stream, xb, wtA, nullptr, nullptr, praw, TOK, HH*PP*3, DD, DD, DD, HH*PP*3,
          0,0,0,0,0,0, 1,1, 1,0);
    k_rotsq<<<(TOK*HH+255)/256, 256, 0, stream>>>(Rb, praw, p_t, qpg, sqq, 0);
    trW(stream, Wpk_l, wtA, DD, HH*PP*3, DD, HH*PP*3);
    gemmF(stream, xb, wtA, nullptr, nullptr, praw, TOK, HH*PP*3, DD, DD, DD, HH*PP*3,
          0,0,0,0,0,0, 1,1, 1,0);
    k_rotsq<<<(TOK*HH+255)/256, 256, 0, stream>>>(Rb, praw, p_t, kpg, sqk, 0);
    trW(stream, Wpv_l, wtA, DD, HH*PP*3, DD, HH*PP*3);
    gemmF(stream, xb, wtA, nullptr, nullptr, praw, TOK, HH*PP*3, DD, DD, DD, HH*PP*3,
          0,0,0,0,0,0, 1,1, 1,0);
    k_rotsq<<<(TOK*HH+255)/256, 256, 0, stream>>>(Rb, praw, p_t, vpgT, nullptr, 1);

    // pair bias: WbT (16,64), then (N*L*L,64)@WbT^T -> (N*L*L,16)
    trW(stream, Wb_l, WbT, PDIM, HH, PDIM, 16);
    gemmF(stream, pair, WbT, nullptr, nullptr, biasb, NB*LL*LL, 16, PDIM,
          PDIM, PDIM, 16, 0,0,0,0,0,0, 1,1, 1,0);

    // qk logits per (n,h): (L,32)@(32,L)^T
    gemmF(stream, qb, kbuf, nullptr, nullptr, Sb, LL, LL, DKH,
          HH*DKH, HH*DKH, LL, sQn, DKH, sQn, DKH, sSn, (long)LL*LL,
          NB*HH, HH, 1,0);
    // point cross per (n,h): (L,32)@(32,L)^T (padded K)
    gemmF(stream, qpg, kpg, nullptr, nullptr, Crb, LL, LL, PPAD,
          HH*PPAD, HH*PPAD, LL, sPn, PPAD, sPn, PPAD, sSn, (long)LL*LL,
          NB*HH, HH, 1,0);

    k_softmax<<<NB*HH*LL, 256, 0, stream>>>(Sb, Crb, biasb, sqq, sqk, gamma, mres, l);

    // o_s per (n,h): (L,L)@vT^T  (B transposed-contiguous, ldb=TOK)
    gemmF(stream, Sb, vT, nullptr, nullptr, osb, LL, DKH, LL,
          LL, TOK, HH*DKH, sSn, (long)LL*LL, (long)LL, (long)DKH*TOK, sQn, DKH,
          NB*HH, HH, 1,0);
    // o_p per (n,i): (16,L)@(L,64) (padded M; B = pair, non-trans)
    gemmF(stream, Sb, pair, nullptr, nullptr, opb, 16, PDIM, LL,
          (long)LL*LL, PDIM, PDIM,
          sSn, LL, (long)LL*LL*PDIM, (long)LL*PDIM, (long)LL*16*PDIM, (long)16*PDIM,
          NB*LL, LL, 0,0);
    // o_ptg per (n,h): (L,L)@vpgT^T (padded N)
    gemmF(stream, Sb, vpgT, nullptr, nullptr, optg, LL, PPAD, LL,
          LL, TOK, HH*PPAD, sSn, (long)LL*LL, (long)LL, (long)PPAD*TOK, sPn, PPAD,
          NB*HH, HH, 1,0);

    k_feat<<<(TOK*HH+255)/256, 256, 0, stream>>>(osb, opb, optg, Rb, p_t, featb);

    trW(stream, Wo_l, wtA, 1536, DD, 1536, DD);
    gemmF(stream, featb, wtA, bo + (size_t)l*DD, mres, f1, TOK, DD, 1536,
          1536, 1536, DD, 0,0,0,0,0,0, 1,1, 1,0);
    k_addln<<<TOK, 128, 0, stream>>>(xb, f1, ln1g + (size_t)l*DD, ln1b + (size_t)l*DD);
    trW(stream, Wt1 + (size_t)l*DD*DD, wtA, DD, DD, DD, DD);
    gemmF(stream, xb, wtA, bt1 + (size_t)l*DD, nullptr, f1, TOK, DD, DD,
          DD, DD, DD, 0,0,0,0,0,0, 1,1, 1,1);
    trW(stream, Wt2 + (size_t)l*DD*DD, wtA, DD, DD, DD, DD);
    gemmF(stream, f1, wtA, bt2 + (size_t)l*DD, nullptr, f2, TOK, DD, DD,
          DD, DD, DD, 0,0,0,0,0,0, 1,1, 1,0);
    k_addln<<<TOK, 128, 0, stream>>>(xb, f2, ln2g + (size_t)l*DD, ln2b + (size_t)l*DD);
  }

  k_infeat<<<(TOK+63)/64, 64, 0, stream>>>(tree, xb, beta, infb);
  float* houts[3] = {ocrd, orot, oseq};
  int    hdout[3] = {3, 3, 20};
  for (int hd = 0; hd < 3; ++hd) {
    trW(stream, head_w[hd][0], wtA, 387, 256, INFP, 256);       // (256, 416) zero-padded
    gemmF(stream, infb, wtA, head_b[hd][0], nullptr, h1, TOK, 256, INFP,
          INFP, INFP, 256, 0,0,0,0,0,0, 1,1, 1,1);
    trW(stream, head_w[hd][1], wtA, 256, 128, 256, 128);
    gemmF(stream, h1, wtA, head_b[hd][1], nullptr, h2, TOK, 128, 256,
          256, 256, 128, 0,0,0,0,0,0, 1,1, 1,1);
    trW(stream, head_w[hd][2], wtA, 128, hdout[hd], 128, 32);   // (32, 128) zero-padded
    gemmF(stream, h2, wtA, nullptr, nullptr, houts[hd], TOK, 32, 128,
          128, 128, 32, 0,0,0,0,0,0, 1,1, 1,0);
  }
  k_final<<<(TOK+63)/64, 64, 0, stream>>>(Rb, v_t, ocrd, orot, oseq,
                                          head_b[0][2], head_b[1][2], head_b[2][2],
                                          mgen, (float*)d_out);
}
